// CognitionModel_41334765256794
// MI455X (gfx1250) — matrix operand order repaired and checked by reading, not compiled, not run on hardware
//
#include <hip/hip_runtime.h>
#include <hip/hip_bf16.h>
#include <math.h>

typedef __bf16 bf16_t;
typedef __attribute__((ext_vector_type(16))) __bf16 v16bf;
typedef __attribute__((ext_vector_type(8)))  float  v8f;
typedef unsigned int u32x4 __attribute__((ext_vector_type(4)));
typedef int          i32x4 __attribute__((ext_vector_type(4)));
typedef int          i32x8 __attribute__((ext_vector_type(8)));

#define B_  4
#define T_  4096
#define D_  256
#define KQ_ 32
#define V_  8
#define L_  4
#define BT_ (B_*T_)

// ---------------- WMMA helpers (CDNA5 16x16x32 bf16, wave32) ----------------

static __device__ __forceinline__ v8f wmma_bf16(v16bf a, v16bf b, v8f c) {
    // (neg_a, A, neg_b, B, c_mod, C, reuse_a, reuse_b)
    return __builtin_amdgcn_wmma_f32_16x16x32_bf16(false, a, false, b, (short)0, c, false, false);
}

// A fragment: 16x32 tile from row-major bf16 matrix (p pre-offset to row0,k0).
// Lanes 0-15: rows 0-15, K halves 0-7 / 16-23 ; lanes 16-31: K halves 8-15 / 24-31.
static __device__ __forceinline__ v16bf load_a(const bf16_t* p0, int ld, int lane) {
    const bf16_t* p = p0 + (lane & 15) * ld + ((lane >> 4) << 3);
    v16bf f;
#pragma unroll
    for (int i = 0; i < 8; ++i) { f[i] = p[i]; f[i + 8] = p[i + 16]; }
    return f;
}
// B fragment from N-major (transposed) matrix Bt[n][k] (p pre-offset to col0,k0).
// Lane = column N; lanes 0-15 hold K 0-7 / 16-23, lanes 16-31 hold K 8-15 / 24-31.
static __device__ __forceinline__ v16bf load_bt(const bf16_t* p0, int ld, int lane) {
    const bf16_t* p = p0 + (lane & 15) * ld + ((lane >> 4) << 3);
    v16bf f;
#pragma unroll
    for (int i = 0; i < 8; ++i) { f[i] = p[i]; f[i + 8] = p[i + 16]; }
    return f;
}

// ---------------- Tensor Data Mover: stage a 2D bf16 tile into LDS ----------------
// D# per CDNA5 ISA: group0 = {count/flags, lds_addr,
// global_addr[31:0], global_addr[56:32] | type=2<<30}; group1 packs data_size=2B,
// tensor_dim0/1, tile_dim0/1, tensor_dim0_stride. Groups 2/3 zero (2D tensor).
static __device__ __forceinline__ void tdm_load_2d_bf16(const bf16_t* gsrc,
                                                        unsigned lds_byte_addr,
                                                        int rows, int cols, int ld) {
    unsigned long long ga = (unsigned long long)gsrc;
    u32x4 g0;
    g0[0] = 1u;                                        // count=1, user descriptor
    g0[1] = lds_byte_addr;                             // lds_addr (bytes)
    g0[2] = (unsigned)(ga & 0xFFFFFFFFu);              // global_addr[31:0]
    g0[3] = (unsigned)((ga >> 32) & 0x1FFFFFFu) | (2u << 30);  // addr[56:32] | type=2
    i32x8 g1;
    g1[0] = (1 << 16);                                 // workgroup_mask=0, data_size=1 (2B)
    g1[1] = (cols & 0xFFFF) << 16;                     // tensor_dim0[15:0] @ bits63:48
    g1[2] = ((cols >> 16) & 0xFFFF) | ((rows & 0xFFFF) << 16);  // dim0 hi | tensor_dim1 lo
    g1[3] = ((rows >> 16) & 0xFFFF) | ((cols & 0xFFFF) << 16);  // dim1 hi | tile_dim0
    g1[4] = (rows & 0xFFFF);                           // tile_dim1 | tile_dim2=0
    g1[5] = ld;                                        // tensor_dim0_stride[31:0]
    g1[6] = 0;                                         // stride hi | dim1_stride lo
    g1[7] = 0;
    i32x4 gz = {0, 0, 0, 0};
#if defined(__clang_major__) && (__clang_major__ >= 23)
    i32x8 gz8 = {0, 0, 0, 0, 0, 0, 0, 0};
    __builtin_amdgcn_tensor_load_to_lds(g0, g1, gz, gz, gz8, 0);
#else
    __builtin_amdgcn_tensor_load_to_lds(g0, g1, gz, gz, 0);
#endif
}

// ---------------- small utility kernels ----------------

__global__ void init_buffers(float* hsum, unsigned* maxbuf) {
    int i = blockIdx.x * blockDim.x + threadIdx.x;
    if (i < B_ * D_) { hsum[i] = 0.f; maxbuf[i] = 0u; }
}

__global__ void cvt_bf16(const float* __restrict__ src, bf16_t* __restrict__ dst, int n) {
    int i = blockIdx.x * blockDim.x + threadIdx.x;
    if (i < n) dst[i] = (bf16_t)src[i];
}

// batched transpose-convert: dst[batch][c][r] = src[batch][r][c]
__global__ void cvt_bf16_T(const float* __restrict__ src, bf16_t* __restrict__ dst,
                           int R, int C) {
    int i = blockIdx.x * blockDim.x + threadIdx.x;
    long base = (long)blockIdx.y * R * C;
    if (i < R * C) {
        int r = i / C, c = i % C;
        dst[base + (long)c * R + r] = (bf16_t)src[base + i];
    }
}

// ---------------- generic WMMA GEMM: C(bf16) = act(A @ Bt^T + bias) ----------------
// A: [M x K] row-major bf16; Bt: [N x K] (N-major) bf16; C: [M x N] bf16.
// ACT: 0 = none, 1 = exact GELU, 2 = ReLU.  One wave per block, NT col-tiles.
template <int NT, int ACT>
__global__ void gemm_bias_act(const bf16_t* __restrict__ A, const bf16_t* __restrict__ Bt,
                              const float* __restrict__ bias, bf16_t* __restrict__ C,
                              int N, int K) {
    const int lane = threadIdx.x;
    const int row0 = blockIdx.x * 16;
    const int col0 = blockIdx.y * 16 * NT;
    v8f acc[NT] = {};
    for (int k0 = 0; k0 < K; k0 += 32) {
        if (k0 + 32 < K) __builtin_prefetch(A + (long)row0 * K + k0 + 32, 0, 0);
        v16bf a = load_a(A + (long)row0 * K + k0, K, lane);
#pragma unroll
        for (int t = 0; t < NT; ++t) {
            v16bf b = load_bt(Bt + (long)(col0 + t * 16) * K + k0, K, lane);
            acc[t] = wmma_bf16(a, b, acc[t]);
        }
    }
    const int rr = (lane >> 4) << 3;
    const int cc = lane & 15;
#pragma unroll
    for (int t = 0; t < NT; ++t) {
        int col = col0 + t * 16 + cc;
        float bv = bias ? bias[col] : 0.f;
#pragma unroll
        for (int j = 0; j < 8; ++j) {
            float v = acc[t][j] + bv;
            if (ACT == 1) v = 0.5f * v * (1.f + erff(v * 0.70710678118654752f));
            else if (ACT == 2) v = fmaxf(v, 0.f);
            C[(long)(row0 + rr + j) * N + col] = (bf16_t)v;
        }
    }
}

// ---------------- router: sim tiles via WMMA + streaming top-4 ----------------
// One wave handles 16 query rows; loops all 4096 key cols in 16-wide WMMA tiles.
// gates>0 and 1/sqrt(KQ)>0 preserve top-k order -> omitted (indices only matter).
__global__ void router_topk(const bf16_t* __restrict__ qb, const bf16_t* __restrict__ kb,
                            const float* __restrict__ mask, int* __restrict__ idx) {
    __shared__ float smask[T_];
    __shared__ float simt[16][16];
    const int lane = threadIdx.x;
    const int row0 = blockIdx.x * 16;        // global row in [0, B*T), 256 blocks per batch
    const int b = row0 / T_;
    for (int i = lane; i < T_; i += 32) smask[i] = mask[b * T_ + i];
    __syncthreads();
    v16bf a = load_a(qb + (long)row0 * KQ_, KQ_, lane);   // K=32: one fragment
    float tv[4] = {-3e38f, -3e38f, -3e38f, -3e38f};
    int   ti[4] = {0, 0, 0, 0};
    const int r = lane;                                   // lanes 0..15 own rows
    const float mt = (lane < 16) ? mask[row0 + r] : 0.f;
    for (int s0 = 0; s0 < T_; s0 += 16) {
        v16bf bf = load_bt(kb + (long)(b * T_ + s0) * KQ_, KQ_, lane); // k is naturally N-major
        v8f acc = {};
        acc = wmma_bf16(a, bf, acc);
        const int rr = (lane >> 4) << 3, cc = lane & 15;
#pragma unroll
        for (int j = 0; j < 8; ++j) simt[rr + j][cc] = acc[j];
        __syncthreads();
        if (lane < 16) {
#pragma unroll
            for (int c = 0; c < 16; ++c) {
                int s = s0 + c;
                float v = (mt != 0.f && smask[s] != 0.f) ? simt[r][c] : -1e9f;
                if (v > tv[3]) {                       // strict > : lowest-index ties win
                    if (v > tv[0])      { tv[3]=tv[2]; ti[3]=ti[2]; tv[2]=tv[1]; ti[2]=ti[1]; tv[1]=tv[0]; ti[1]=ti[0]; tv[0]=v; ti[0]=s; }
                    else if (v > tv[1]) { tv[3]=tv[2]; ti[3]=ti[2]; tv[2]=tv[1]; ti[2]=ti[1]; tv[1]=v; ti[1]=s; }
                    else if (v > tv[2]) { tv[3]=tv[2]; ti[3]=ti[2]; tv[2]=v; ti[2]=s; }
                    else                { tv[3]=v; ti[3]=s; }
                }
            }
        }
        __syncthreads();
    }
    if (lane < 16) {
#pragma unroll
        for (int j = 0; j < 4; ++j) idx[(row0 + r) * 4 + j] = ti[j];
    }
}

// routed_mean = mean of 4 gathered x rows (kept f32 for the residual, bf16 for GEMM)
__global__ void gather_mean(const float* __restrict__ x, const int* __restrict__ idx,
                            float* __restrict__ rf, bf16_t* __restrict__ rb) {
    int i = blockIdx.x * blockDim.x + threadIdx.x;
    if (i >= BT_ * D_) return;
    int row = i / D_, d = i % D_;
    int b = row / T_;
    const int* id = idx + row * 4;
    const float* xb = x + (long)b * T_ * D_ + d;
    float s = 0.25f * (xb[(long)id[0] * D_] + xb[(long)id[1] * D_] +
                       xb[(long)id[2] * D_] + xb[(long)id[3] * D_]);
    rf[i] = s;
    rb[i] = (bf16_t)s;
}

// ---------------- FFN2 + residual + LayerNorm + h-mean partials ----------------
// Block = 8 waves; 16 rows x full 256 cols so LN stays on-chip.
__global__ void ffn2_ln(const bf16_t* __restrict__ H1, const bf16_t* __restrict__ W2t,
                        const float* __restrict__ b2, const float* __restrict__ resid,
                        const float* __restrict__ g, const float* __restrict__ beta,
                        bf16_t* __restrict__ Hout, float* __restrict__ hsum) {
    __shared__ float sm[16][D_];
    __shared__ float smu[16], srv[16];
    const int tid = threadIdx.x, lane = tid & 31, wave = tid >> 5;
    const int row0 = blockIdx.x * 16;
    const int b = row0 / T_;
    v8f acc[2] = {};
    for (int k0 = 0; k0 < 2 * D_; k0 += 32) {
        if (k0 + 32 < 2 * D_) __builtin_prefetch(H1 + (long)row0 * (2 * D_) + k0 + 32, 0, 0);
        v16bf a = load_a(H1 + (long)row0 * (2 * D_) + k0, 2 * D_, lane);
#pragma unroll
        for (int t = 0; t < 2; ++t) {
            v16bf bf = load_bt(W2t + (long)(wave * 32 + t * 16) * (2 * D_) + k0, 2 * D_, lane);
            acc[t] = wmma_bf16(a, bf, acc[t]);
        }
    }
    const int rr = (lane >> 4) << 3, cc = lane & 15;
#pragma unroll
    for (int t = 0; t < 2; ++t) {
        int col = wave * 32 + t * 16 + cc;
#pragma unroll
        for (int j = 0; j < 8; ++j) {
            int row = rr + j;
            sm[row][col] = acc[t][j] + b2[col] + resid[(long)(row0 + row) * D_ + col];
        }
    }
    __syncthreads();
    if (tid < 16) {
        float s = 0.f, s2 = 0.f;
        for (int c = 0; c < D_; ++c) s += sm[tid][c];
        float mu = s * (1.f / D_);
        for (int c = 0; c < D_; ++c) { float d = sm[tid][c] - mu; s2 += d * d; }
        smu[tid] = mu;
        srv[tid] = rsqrtf(s2 * (1.f / D_) + 1e-5f);
    }
    __syncthreads();
    {
        const int row = tid >> 4, cb = tid & 15;
        float mu = smu[row], rv = srv[row];
#pragma unroll
        for (int i = 0; i < 16; ++i) {
            int c = cb + i * 16;
            float v = (sm[row][c] - mu) * rv * g[c] + beta[c];
            Hout[(long)(row0 + row) * D_ + c] = (bf16_t)v;
            sm[row][c] = v;
        }
    }
    __syncthreads();
    {
        int c = tid;
        float s = 0.f;
#pragma unroll
        for (int r2 = 0; r2 < 16; ++r2) s += sm[r2][c];
        atomicAdd(&hsum[b * D_ + c], s);      // 256 partial-sum atomics per column
    }
}

// ---------------- tiny GRU + program logits + softmax-mean weights ----------------
__global__ void gru_program(const float* __restrict__ hsum, const float* __restrict__ Whh,
                            const float* __restrict__ b_ih, const float* __restrict__ b_hh,
                            const float* __restrict__ Wp, const float* __restrict__ bp,
                            float* __restrict__ wout) {
    __shared__ float hs[B_][D_];
    __shared__ float gh[B_][3 * D_];
    __shared__ float lg[B_][L_][V_];
    const int tid = threadIdx.x;  // 256
    for (int i = tid; i < B_ * D_; i += 256) hs[i / D_][i % D_] = hsum[i] * (1.f / T_);
    __syncthreads();
    for (int l = 0; l < L_; ++l) {
        for (int o = tid; o < B_ * 3 * D_; o += 256) {
            int b = o / (3 * D_), oo = o % (3 * D_);
            const float* wr = Whh + (long)oo * D_;
            float s = 0.f;
            for (int d = 0; d < D_; ++d) s += hs[b][d] * wr[d];
            gh[b][oo] = s + b_hh[oo];
        }
        __syncthreads();
        for (int i = tid; i < B_ * D_; i += 256) {
            int b = i / D_, d = i % D_;
            float r = 1.f / (1.f + expf(-(b_ih[d] + gh[b][d])));
            float z = 1.f / (1.f + expf(-(b_ih[D_ + d] + gh[b][D_ + d])));
            float n = tanhf(b_ih[2 * D_ + d] + r * gh[b][2 * D_ + d]);
            hs[b][d] = (1.f - z) * n + z * hs[b][d];
        }
        __syncthreads();
        if (tid < B_ * V_) {
            int b = tid / V_, v = tid % V_;
            float s = bp[v];
            for (int d = 0; d < D_; ++d) s += hs[b][d] * Wp[d * V_ + v];
            lg[b][l][v] = s;
        }
        __syncthreads();
    }
    if (tid < B_ * V_) {
        int b = tid / V_, v = tid % V_;
        float acc = 0.f;
        for (int l = 0; l < L_; ++l) {
            float m = -3e38f;
            for (int vv = 0; vv < V_; ++vv) m = fmaxf(m, lg[b][l][vv]);
            float den = 0.f;
            for (int vv = 0; vv < V_; ++vv) den += expf(lg[b][l][vv] - m);
            acc += expf(lg[b][l][v] - m) / den;
        }
        wout[b * V_ + v] = acc * (1.f / L_);
    }
}

// ---------------- fused NeuralOpLibrary (dominant 34 GF) + column max ----------------
// Block: 8 waves, 32 rows x 256 cols. The H tile (32x256 bf16 = 16 KB) is staged
// into LDS ONCE via the Tensor Data Mover (TENSORcnt + barrier); both GEMM phases
// then feed A-fragments from LDS. Per v: hidden tile bf16 in LDS, fused f32 in LDS
// scaled by w[b,v]; epilogue is a monotone-uint atomic column max.
__global__ void neural_ops(const bf16_t* __restrict__ H, const bf16_t* __restrict__ W1t,
                           const float* __restrict__ b1, const bf16_t* __restrict__ W2t,
                           const float* __restrict__ b2, const float* __restrict__ wv,
                           unsigned* __restrict__ maxbuf) {
    __shared__ bf16_t hin[32][D_];   // 16 KB : TDM-staged input tile
    __shared__ bf16_t hid[32][D_];   // 16 KB : per-v hidden tile
    __shared__ float  fus[32][D_];   // 32 KB : fused accumulator
    const int tid = threadIdx.x, lane = tid & 31, wave = tid >> 5;
    const int row0 = blockIdx.x * 32;
    const int b = row0 / T_;
    if (wave == 0) {
        // one TDM descriptor per block: 32x256 bf16 tile, row stride 256 elements
        tdm_load_2d_bf16(H + (long)row0 * D_, (unsigned)(unsigned long long)(&hin[0][0]),
                         32, D_, D_);
        __builtin_amdgcn_s_wait_tensorcnt(0);
    }
    for (int i = tid; i < 32 * D_; i += 256) fus[i >> 8][i & 255] = 0.f;
    __syncthreads();
    for (int v = 0; v < V_; ++v) {
        const bf16_t* w1 = W1t + (long)v * D_ * D_;
        const bf16_t* w2 = W2t + (long)v * D_ * D_;
        const float wb = wv[b * V_ + v];
        // Phase A: hidden = relu(H_tile @ opW1[v] + b1[v])  (A from LDS)
#pragma unroll
        for (int t = 0; t < 4; ++t) {
            int tile = wave + t * 8;          // 32 tiles = 2 mtiles x 16 ntiles
            int mt = tile & 1, nt = tile >> 1;
            v8f acc = {};
            for (int k0 = 0; k0 < D_; k0 += 32) {
                if (k0 + 32 < D_) __builtin_prefetch(w1 + (long)(nt * 16) * D_ + k0 + 32, 0, 0);
                v16bf a  = load_a((const bf16_t*)&hin[mt * 16][0] + k0, D_, lane);
                v16bf bf = load_bt(w1 + (long)(nt * 16) * D_ + k0, D_, lane);
                acc = wmma_bf16(a, bf, acc);
            }
            const int rr = (lane >> 4) << 3, cc = lane & 15;
#pragma unroll
            for (int j = 0; j < 8; ++j) {
                int col = nt * 16 + cc;
                float val = fmaxf(acc[j] + b1[v * D_ + col], 0.f);
                hid[mt * 16 + rr + j][col] = (bf16_t)val;
            }
        }
        __syncthreads();
        // Phase B: fused += w[b,v] * (hidden @ opW2[v] + b2[v])  (A from LDS)
#pragma unroll
        for (int t = 0; t < 4; ++t) {
            int tile = wave + t * 8;
            int mt = tile & 1, nt = tile >> 1;
            v8f acc = {};
            for (int k0 = 0; k0 < D_; k0 += 32) {
                if (k0 + 32 < D_) __builtin_prefetch(w2 + (long)(nt * 16) * D_ + k0 + 32, 0, 0);
                v16bf a  = load_a((const bf16_t*)&hid[mt * 16][0] + k0, D_, lane);
                v16bf bf = load_bt(w2 + (long)(nt * 16) * D_ + k0, D_, lane);
                acc = wmma_bf16(a, bf, acc);
            }
            const int rr = (lane >> 4) << 3, cc = lane & 15;
#pragma unroll
            for (int j = 0; j < 8; ++j) {
                int col = nt * 16 + cc;
                fus[mt * 16 + rr + j][col] += wb * (acc[j] + b2[v * D_ + col]);
            }
        }
        __syncthreads();
    }
    {   // column max over the 32 rows -> monotone-uint atomic max
        int c = tid;
        float m = -3e38f;
#pragma unroll
        for (int r2 = 0; r2 < 32; ++r2) m = fmaxf(m, fus[r2][c]);
        unsigned u = __float_as_uint(m);
        u = (u & 0x80000000u) ? ~u : (u | 0x80000000u);
        atomicMax(&maxbuf[b * D_ + c], u);
    }
}

__global__ void out_proj(const unsigned* __restrict__ maxbuf, const float* __restrict__ Wo,
                         const float* __restrict__ bo, float* __restrict__ out) {
    __shared__ float pooled[D_];
    const int b = blockIdx.x, tid = threadIdx.x;  // 256 threads
    unsigned u = maxbuf[b * D_ + tid];
    pooled[tid] = (u & 0x80000000u) ? __uint_as_float(u ^ 0x80000000u) : __uint_as_float(~u);
    __syncthreads();
    float s = bo[tid];
    for (int d = 0; d < D_; ++d) s += pooled[d] * Wo[d * D_ + tid];
    out[b * D_ + tid] = s;
}

// ---------------- launch ----------------

extern "C" void kernel_launch(void* const* d_in, const int* in_sizes, int n_in,
                              void* d_out, int out_size, void* d_ws, size_t ws_size,
                              hipStream_t stream) {
    (void)in_sizes; (void)n_in; (void)out_size; (void)ws_size;
    const float* x   = (const float*)d_in[0];
    const float* am  = (const float*)d_in[1];
    const float* Wq  = (const float*)d_in[2];
    const float* bq  = (const float*)d_in[3];
    const float* Wk  = (const float*)d_in[4];
    const float* bk  = (const float*)d_in[5];
    // d_in[6] Wg, d_in[7] bg: positive per-row scale, cannot change top-k -> unused
    const float* W1f = (const float*)d_in[8];
    const float* b1f = (const float*)d_in[9];
    const float* W2f = (const float*)d_in[10];
    const float* b2f = (const float*)d_in[11];
    const float* lng = (const float*)d_in[12];
    const float* lnb = (const float*)d_in[13];
    const float* Whh = (const float*)d_in[14];
    const float* bih = (const float*)d_in[15];
    const float* bhh = (const float*)d_in[16];
    const float* Wp  = (const float*)d_in[17];
    const float* bp  = (const float*)d_in[18];
    const float* oW1 = (const float*)d_in[19];
    const float* ob1 = (const float*)d_in[20];
    const float* oW2 = (const float*)d_in[21];
    const float* ob2 = (const float*)d_in[22];
    const float* Wo  = (const float*)d_in[23];
    const float* bo  = (const float*)d_in[24];

    char* ws = (char*)d_ws;
    size_t off = 0;
    auto alloc = [&](size_t bytes) { void* p = ws + off; off = (off + bytes + 255) & ~size_t(255); return p; };
    bf16_t*   x_bf = (bf16_t*)alloc((size_t)BT_ * D_ * 2);
    bf16_t*   q_bf = (bf16_t*)alloc((size_t)BT_ * KQ_ * 2);
    bf16_t*   k_bf = (bf16_t*)alloc((size_t)BT_ * KQ_ * 2);
    bf16_t*   WqT  = (bf16_t*)alloc((size_t)KQ_ * D_ * 2);
    bf16_t*   WkT  = (bf16_t*)alloc((size_t)KQ_ * D_ * 2);
    bf16_t*   W1fT = (bf16_t*)alloc((size_t)2 * D_ * D_ * 2);
    bf16_t*   W2fT = (bf16_t*)alloc((size_t)D_ * 2 * D_ * 2);
    bf16_t*   oW1T = (bf16_t*)alloc((size_t)V_ * D_ * D_ * 2);
    bf16_t*   oW2T = (bf16_t*)alloc((size_t)V_ * D_ * D_ * 2);
    int*      idx  = (int*)alloc((size_t)BT_ * 4 * 4);
    float*    rf   = (float*)alloc((size_t)BT_ * D_ * 4);
    bf16_t*   rb   = (bf16_t*)alloc((size_t)BT_ * D_ * 2);
    bf16_t*   h1   = (bf16_t*)alloc((size_t)BT_ * 2 * D_ * 2);
    bf16_t*   hb   = (bf16_t*)alloc((size_t)BT_ * D_ * 2);
    float*    hsum = (float*)alloc((size_t)B_ * D_ * 4);
    float*    wbuf = (float*)alloc((size_t)B_ * V_ * 4);
    unsigned* maxb = (unsigned*)alloc((size_t)B_ * D_ * 4);

    init_buffers<<<(B_ * D_ + 255) / 256, 256, 0, stream>>>(hsum, maxb);
    cvt_bf16<<<(BT_ * D_ + 255) / 256, 256, 0, stream>>>(x, x_bf, BT_ * D_);
    cvt_bf16_T<<<dim3((D_ * KQ_ + 255) / 256, 1), 256, 0, stream>>>(Wq, WqT, D_, KQ_);
    cvt_bf16_T<<<dim3((D_ * KQ_ + 255) / 256, 1), 256, 0, stream>>>(Wk, WkT, D_, KQ_);
    cvt_bf16_T<<<dim3((D_ * 2 * D_ + 255) / 256, 1), 256, 0, stream>>>(W1f, W1fT, D_, 2 * D_);
    cvt_bf16_T<<<dim3((2 * D_ * D_ + 255) / 256, 1), 256, 0, stream>>>(W2f, W2fT, 2 * D_, D_);
    cvt_bf16_T<<<dim3((D_ * D_ + 255) / 256, V_), 256, 0, stream>>>(oW1, oW1T, D_, D_);
    cvt_bf16_T<<<dim3((D_ * D_ + 255) / 256, V_), 256, 0, stream>>>(oW2, oW2T, D_, D_);

    // q/k projections: [BT x 256] @ [256 x 32]
    gemm_bias_act<2, 0><<<dim3(BT_ / 16, 1), 32, 0, stream>>>(x_bf, WqT, bq, q_bf, KQ_, D_);
    gemm_bias_act<2, 0><<<dim3(BT_ / 16, 1), 32, 0, stream>>>(x_bf, WkT, bk, k_bf, KQ_, D_);
    // router: WMMA sim tiles + streaming top-4, then gather-mean
    router_topk<<<BT_ / 16, 32, 0, stream>>>(q_bf, k_bf, am, idx);
    gather_mean<<<(BT_ * D_ + 255) / 256, 256, 0, stream>>>(x, idx, rf, rb);
    // FFN1 + exact GELU: [BT x 256] @ [256 x 512]
    gemm_bias_act<4, 1><<<dim3(BT_ / 16, (2 * D_) / 64), 32, 0, stream>>>(rb, W1fT, b1f, h1, 2 * D_, D_);
    // FFN2 + residual + LayerNorm + h-mean partials
    ffn2_ln<<<BT_ / 16, 256, 0, stream>>>(h1, W2fT, b2f, rf, lng, lnb, hb, hsum);
    // GRU + program logits -> softmax-mean weights
    gru_program<<<1, 256, 0, stream>>>(hsum, Whh, bih, bhh, Wp, bp, wbuf);
    // fused op library (TDM-staged) + max pool
    neural_ops<<<BT_ / 32, 256, 0, stream>>>(hb, oW1T, ob1, oW2T, ob2, wbuf, maxb);
    // output projection
    out_proj<<<B_, 256, 0, stream>>>(maxb, Wo, bo, (float*)d_out);
}
